// TransformerEncoder_67697274519855
// MI455X (gfx1250) — matrix operand order repaired and checked by reading, not compiled, not run on hardware
//
#include <hip/hip_runtime.h>
#include <hip/hip_bf16.h>

typedef __attribute__((ext_vector_type(16))) __bf16 v16bf;
typedef __attribute__((ext_vector_type(8)))  float  v8f;
typedef __attribute__((ext_vector_type(4)))  unsigned int ui32x4;
typedef __attribute__((ext_vector_type(8)))  int i32x8;
typedef __attribute__((ext_vector_type(4)))  int i32x4;

#define TM 64
#define TN 128
#define KC 32

// ---------------------------------------------------------------------------
// TDM: async 2D tile load (bf16 elements) Global -> LDS via Tensor Data Mover.
// tile_x = contiguous elements per row, tile_y = rows, stride = row stride
// (elements). LDS receives the tile packed: lds[row*tile_x + x].
// D# layout per CDNA5 ISA (group0 + group1,
// groups 2/3 zero for a 2D tile). 6-arg builtin form (clang-23 toolchain).
// ---------------------------------------------------------------------------
__device__ __forceinline__ void tdm_load_2d_bf16(unsigned lds_off, const void* gaddr,
                                                 unsigned tile_x, unsigned tile_y,
                                                 unsigned stride_elems)
{
    unsigned long long ga = (unsigned long long)(uintptr_t)gaddr;
    ui32x4 g0;
    g0[0] = 1u;                                              // count=1, user desc
    g0[1] = lds_off;                                         // lds_addr (bytes)
    g0[2] = (unsigned)(ga & 0xffffffffu);                    // global_addr[31:0]
    g0[3] = (unsigned)((ga >> 32) & 0x01ffffffu) | 0x80000000u; // addr[56:32] | type=2
    i32x8 g1;
    g1[0] = (int)(1u << 16);                                 // data_size=1 (2 bytes)
    g1[1] = (int)((tile_x & 0xffffu) << 16);                 // tensor_dim0 = tile_x
    g1[2] = (int)((tile_y & 0xffffu) << 16);                 // tensor_dim1 = tile_y
    g1[3] = (int)((tile_x & 0xffffu) << 16);                 // tile_dim0
    g1[4] = (int)(tile_y & 0xffffu);                         // tile_dim1 (tile_dim2=0)
    g1[5] = (int)stride_elems;                               // tensor_dim0_stride lo32
    g1[6] = 0;
    g1[7] = 0;
    i32x4 gz4 = {0, 0, 0, 0};
    i32x8 gz8 = {0, 0, 0, 0, 0, 0, 0, 0};
    __builtin_amdgcn_tensor_load_to_lds(g0, g1, gz4, gz4, gz8, 0);
}

// ---------------------------------------------------------------------------
// GEMM: C[M,N] = act(A[M,K] @ W + bias[N]); A fp32 row-major, weight given as
// bf16 TRANSPOSED Wt[N][K]. bf16 WMMA core, fp32 accumulate.
// block = 256 threads (8 waves, 2x4), 64x128 tile, wave = 32x32 (2x2 WMMA).
// Weight tiles (32k x 128n, stored [col][k] in LDS) are fetched by the TDM
// with double buffering: tile i+1 is in flight while tile i is consumed.
// Requires M%64==0, N%128==0, K%32==0.
// ---------------------------------------------------------------------------
__global__ __launch_bounds__(256) void gemm_wt(
    const float* __restrict__ A, const __bf16* __restrict__ Wt,
    const float* __restrict__ bias, float* __restrict__ C,
    int M, int N, int K, int relu)
{
    __shared__ __bf16 sA[TM * KC];        // [row][k]  4KB
    __shared__ __bf16 sB[2][TN * KC];     // [col][k]  2 x 8KB (TDM destinations)

    const int tid  = threadIdx.x;
    const int lane = tid & 31;
    const int wave = tid >> 5;
    const int lo   = lane & 15;
    const int hi   = lane >> 4;        // 0 or 1
    const int wrow = wave >> 2;        // 0..1
    const int wcol = wave & 3;         // 0..3
    const int m0   = blockIdx.y * TM;
    const int n0   = blockIdx.x * TN;
    const unsigned sB_off[2] = { (unsigned)(uintptr_t)(void*)&sB[0][0],
                                 (unsigned)(uintptr_t)(void*)&sB[1][0] };

    v8f acc[2][2] = {};
    const int ntiles = K / KC;

    if (wave == 0)   // prologue: DMA first weight tile
        tdm_load_2d_bf16(sB_off[0], Wt + (size_t)n0 * K, KC, TN, (unsigned)K);

    for (int it = 0; it < ntiles; ++it) {
        const int kc = it * KC;
        __syncthreads();  // previous tile fully consumed (sA and sB[(it+1)&1])
        if (wave == 0 && it + 1 < ntiles)   // DMA next tile into other buffer
            tdm_load_2d_bf16(sB_off[(it + 1) & 1],
                             Wt + (size_t)n0 * K + kc + KC, KC, TN, (unsigned)K);
        // stage A tile (64x32) fp32->bf16 with b128 global loads
        for (int t4 = tid; t4 < TM * KC / 4; t4 += 256) {
            int r = t4 >> 3, c4 = (t4 & 7) * 4;
            float4 vv = *(const float4*)(A + (size_t)(m0 + r) * K + kc + c4);
            __bf16* d = &sA[r * KC + c4];
            d[0] = (__bf16)vv.x; d[1] = (__bf16)vv.y;
            d[2] = (__bf16)vv.z; d[3] = (__bf16)vv.w;
        }
        if (wave == 0) {   // tensor ops complete in order: <=1 left => tile it done
            if (it + 1 < ntiles) __builtin_amdgcn_s_wait_tensorcnt(1);
            else                 __builtin_amdgcn_s_wait_tensorcnt(0);
        }
        __syncthreads();

        const __bf16* sBc = sB[it & 1];
        v16bf af[2], bwf[2];
        #pragma unroll
        for (int ti = 0; ti < 2; ++ti) {
            int row = wrow * 32 + ti * 16 + lo;
            #pragma unroll
            for (int e = 0; e < 16; ++e) {
                int kk = (e < 8 ? e : e + 8) + hi * 8;   // contiguous 8-elem runs
                af[ti][e] = sA[row * KC + kk];
            }
        }
        #pragma unroll
        for (int tj = 0; tj < 2; ++tj) {
            int col = wcol * 32 + tj * 16 + lo;
            #pragma unroll
            for (int e = 0; e < 16; ++e)                 // contiguous 8-elem runs
                bwf[tj][e] = sBc[col * KC + (e < 8 ? e : e + 8) + hi * 8];
        }
        #pragma unroll
        for (int ti = 0; ti < 2; ++ti)
            #pragma unroll
            for (int tj = 0; tj < 2; ++tj)
                acc[ti][tj] = __builtin_amdgcn_wmma_f32_16x16x32_bf16(
                    false, af[ti], false, bwf[tj], (short)0, acc[ti][tj], false, false);
    }

    #pragma unroll
    for (int ti = 0; ti < 2; ++ti)
        #pragma unroll
        for (int tj = 0; tj < 2; ++tj) {
            int colb = n0 + wcol * 32 + tj * 16 + lo;
            float bv = bias ? bias[colb] : 0.0f;
            #pragma unroll
            for (int r = 0; r < 8; ++r) {
                int row = m0 + wrow * 32 + ti * 16 + hi * 8 + r;  // C-frag layout
                float v = acc[ti][tj][r] + bv;
                if (relu) v = fmaxf(v, 0.0f);
                C[(size_t)row * N + colb] = v;
            }
        }
}

// ---------------------------------------------------------------------------
// Flash attention (bf16 WMMA, fp32 accumulate, online softmax).
// q,k,v,out: [B*L, D] fp32, head h = cols [h*64, h*64+64).
// grid = (B*H, L/128), block = 256; each wave owns 16 query rows.
// ---------------------------------------------------------------------------
#define VKS 80   // sVt row stride (elems): 160B keeps 16B alignment, spreads banks

__global__ __launch_bounds__(256) void flash_attn(
    const float* __restrict__ q, const float* __restrict__ k,
    const float* __restrict__ v, float* __restrict__ out)
{
    const int Lc = 512, Dc = 1024, Hh = 16, DH = 64, KT = 64;
    const float scale = 0.125f;   // 1/sqrt(64)

    __shared__ __bf16 sK[KT * DH];        // [key][dh]  8KB
    __shared__ __bf16 sVt[DH * VKS];      // [dh][key]  10KB (transposed)
    __shared__ __bf16 sP[8][16 * KT];     // per-wave P tile, 16KB

    const int bh = blockIdx.x;
    const int b  = bh / Hh, h = bh % Hh;
    const int qt = blockIdx.y;

    const int tid  = threadIdx.x;
    const int lane = tid & 31;
    const int wave = tid >> 5;
    const int lo   = lane & 15;
    const int hi   = lane >> 4;

    const int qrow0 = qt * 128 + wave * 16;
    const size_t qbase = ((size_t)b * Lc + qrow0) * Dc + h * DH;

    // Q fragments: A-matrix 16x64 = two 16x32 bf16 fragments
    v16bf qf[2];
    #pragma unroll
    for (int p = 0; p < 2; ++p)
        #pragma unroll
        for (int e = 0; e < 16; ++e) {
            int dh = p * 32 + (e < 8 ? e : e + 8) + hi * 8;
            qf[p][e] = (__bf16)q[qbase + (size_t)lo * Dc + dh];
        }

    v8f O[4] = {};
    float mstate[8], lstate[8];
    #pragma unroll
    for (int r = 0; r < 8; ++r) { mstate[r] = -1e30f; lstate[r] = 0.0f; }

    for (int kt = 0; kt < Lc; kt += KT) {
        __syncthreads();
        for (int t4 = tid; t4 < KT * DH / 4; t4 += 256) {
            int r = t4 >> 4, c4 = (t4 & 15) * 4;
            size_t g = ((size_t)b * Lc + kt + r) * Dc + h * DH + c4;
            float4 kv = *(const float4*)(k + g);
            float4 vv = *(const float4*)(v + g);
            __bf16* dk = &sK[r * DH + c4];
            dk[0] = (__bf16)kv.x; dk[1] = (__bf16)kv.y;
            dk[2] = (__bf16)kv.z; dk[3] = (__bf16)kv.w;
            sVt[(c4 + 0) * VKS + r] = (__bf16)vv.x;
            sVt[(c4 + 1) * VKS + r] = (__bf16)vv.y;
            sVt[(c4 + 2) * VKS + r] = (__bf16)vv.z;
            sVt[(c4 + 3) * VKS + r] = (__bf16)vv.w;
        }
        __syncthreads();

        // S = Q @ K^T : 4 fragments of 16 keys each
        v8f S[4];
        #pragma unroll
        for (int js = 0; js < 4; ++js) {
            v8f a = {};
            #pragma unroll
            for (int p = 0; p < 2; ++p) {
                v16bf bfr;
                #pragma unroll
                for (int e = 0; e < 16; ++e)          // 8-elem runs in dh
                    bfr[e] = sK[(js * 16 + lo) * DH + p * 32 + (e < 8 ? e : e + 8) + hi * 8];
                a = __builtin_amdgcn_wmma_f32_16x16x32_bf16(
                        false, qf[p], false, bfr, (short)0, a, false, false);
            }
            S[js] = a;
        }

        // online softmax per row (row = hi*8 + r); 16-lane halves share a row
        #pragma unroll
        for (int r = 0; r < 8; ++r) {
            float sv0 = S[0][r] * scale, sv1 = S[1][r] * scale;
            float sv2 = S[2][r] * scale, sv3 = S[3][r] * scale;
            float mx = fmaxf(fmaxf(sv0, sv1), fmaxf(sv2, sv3));
            for (int off = 1; off < 16; off <<= 1)
                mx = fmaxf(mx, __shfl_xor(mx, off, 32));
            float mnew  = fmaxf(mstate[r], mx);
            float alpha = __expf(mstate[r] - mnew);
            float p0 = __expf(sv0 - mnew), p1 = __expf(sv1 - mnew);
            float p2 = __expf(sv2 - mnew), p3 = __expf(sv3 - mnew);
            int rr = (hi * 8 + r) * KT + lo;
            sP[wave][rr +  0] = (__bf16)p0;
            sP[wave][rr + 16] = (__bf16)p1;
            sP[wave][rr + 32] = (__bf16)p2;
            sP[wave][rr + 48] = (__bf16)p3;
            float ls = p0 + p1 + p2 + p3;
            for (int off = 1; off < 16; off <<= 1)
                ls += __shfl_xor(ls, off, 32);
            lstate[r] = lstate[r] * alpha + ls;
            mstate[r] = mnew;
            #pragma unroll
            for (int vt = 0; vt < 4; ++vt) O[vt][r] *= alpha;
        }

        // O += P @ V   (P re-read from per-wave LDS in A-fragment layout)
        v16bf pf[2];
        #pragma unroll
        for (int p = 0; p < 2; ++p)
            #pragma unroll
            for (int e = 0; e < 16; ++e) {
                int kk = p * 32 + (e < 8 ? e : e + 8) + hi * 8;  // K-dim = key
                pf[p][e] = sP[wave][lo * KT + kk];
            }
        #pragma unroll
        for (int vt = 0; vt < 4; ++vt)
            #pragma unroll
            for (int p = 0; p < 2; ++p) {
                v16bf bfr;
                #pragma unroll
                for (int e = 0; e < 16; ++e)          // 8-elem runs in key
                    bfr[e] = sVt[(vt * 16 + lo) * VKS + p * 32 + (e < 8 ? e : e + 8) + hi * 8];
                O[vt] = __builtin_amdgcn_wmma_f32_16x16x32_bf16(
                            false, pf[p], false, bfr, (short)0, O[vt], false, false);
            }
    }

    #pragma unroll
    for (int vt = 0; vt < 4; ++vt)
        #pragma unroll
        for (int r = 0; r < 8; ++r) {
            float ov = O[vt][r] / lstate[r];
            out[((size_t)b * Lc + qrow0 + hi * 8 + r) * Dc + h * DH + vt * 16 + lo] = ov;
        }
}

// ---------------------------------------------------------------------------
// Weight convert + transpose: src fp32 [K][N] -> dst bf16 [N][K]
// ---------------------------------------------------------------------------
__global__ void convT_bf16(const float* __restrict__ src, __bf16* __restrict__ dst,
                           int K, int N, size_t total)
{
    size_t i = (size_t)blockIdx.x * blockDim.x + threadIdx.x;
    if (i >= total) return;
    int n = (int)(i / (size_t)K), kk = (int)(i % (size_t)K);
    dst[i] = (__bf16)src[(size_t)kk * N + n];
}

// ---------------------------------------------------------------------------
// BatchNorm (training stats over all M rows), with fused optional residual:
// input is x (+ res if non-null). D fixed to 1024.
// ---------------------------------------------------------------------------
__global__ void bn_stats(const float* __restrict__ x, const float* __restrict__ res,
                         float* __restrict__ stats, int M, int D)
{
    int ch = blockIdx.x * blockDim.x + threadIdx.x;
    float s = 0.0f, s2 = 0.0f;
    for (int r = 0; r < M; ++r) {
        float v = x[(size_t)r * D + ch];
        if (res) v += res[(size_t)r * D + ch];
        s += v; s2 += v * v;
    }
    float mean = s / (float)M;
    float var  = s2 / (float)M - mean * mean;
    stats[ch]     = mean;
    stats[D + ch] = rsqrtf(var + 1e-3f);
}

__global__ void bn_apply(const float* __restrict__ x, const float* __restrict__ res,
                         const float* __restrict__ stats,
                         const float* __restrict__ gamma, const float* __restrict__ beta,
                         float* __restrict__ y, size_t n4)
{
    size_t i = (size_t)blockIdx.x * blockDim.x + threadIdx.x;
    if (i >= n4) return;
    size_t j = i * 4;
    int ch = (int)(j & 1023);                       // D = 1024
    float4 xv = *(const float4*)(x + j);
    if (res) {
        float4 rv = *(const float4*)(res + j);
        xv.x += rv.x; xv.y += rv.y; xv.z += rv.z; xv.w += rv.w;
    }
    float4 g  = *(const float4*)(gamma + ch);
    float4 bt = *(const float4*)(beta + ch);
    float4 mu = *(const float4*)(stats + ch);
    float4 rs = *(const float4*)(stats + 1024 + ch);
    float4 o;
    o.x = g.x * (xv.x - mu.x) * rs.x + bt.x;
    o.y = g.y * (xv.y - mu.y) * rs.y + bt.y;
    o.z = g.z * (xv.z - mu.z) * rs.z + bt.z;
    o.w = g.w * (xv.w - mu.w) * rs.w + bt.w;
    *(float4*)(y + j) = o;
}

// ---------------------------------------------------------------------------
extern "C" void kernel_launch(void* const* d_in, const int* in_sizes, int n_in,
                              void* d_out, int out_size, void* d_ws, size_t ws_size,
                              hipStream_t stream)
{
    const int Bb = 8, Lc = 512, Din = 64, D = 1024, S = 2, H = 16;
    const int M = Bb * Lc;                 // 4096
    const size_t MD = (size_t)M * D;
    const size_t DD = (size_t)D * D;

    const float* x       = (const float*)d_in[0];
    const float* W_embed = (const float*)d_in[1];
    const float* bn1_g   = (const float*)d_in[2];
    const float* bn1_b   = (const float*)d_in[3];
    const float* Wq      = (const float*)d_in[4];
    const float* bq      = (const float*)d_in[5];
    const float* Wk      = (const float*)d_in[6];
    const float* bk      = (const float*)d_in[7];
    const float* Wv      = (const float*)d_in[8];
    const float* bv      = (const float*)d_in[9];
    const float* bng     = (const float*)d_in[10];
    const float* bnb     = (const float*)d_in[11];
    const float* Wc1     = (const float*)d_in[12];
    const float* bc1     = (const float*)d_in[13];
    const float* bn2g    = (const float*)d_in[14];
    const float* bn2b    = (const float*)d_in[15];
    float* out = (float*)d_out;

    float* ws    = (float*)d_ws;
    float* enc   = ws;               // [M,D] fp32
    float* qb    = ws + 1 * MD;
    float* kb    = ws + 2 * MD;
    float* vb    = ws + 3 * MD;
    float* t0    = ws + 4 * MD;
    float* stats = ws + 5 * MD;      // [2*D]
    __bf16* wtE  = (__bf16*)(stats + 2 * D);     // [D][Din]
    __bf16* wtQ  = wtE + (size_t)D * Din;        // S*[D][D] transposed
    __bf16* wtK  = wtQ + (size_t)S * DD;
    __bf16* wtV  = wtK + (size_t)S * DD;
    __bf16* wtC  = wtV + (size_t)S * DD;

    const dim3 blk(256);
    const dim3 ggrid(D / TN, M / TM);         // (8, 64)
    const dim3 agrid(Bb * H, Lc / 128);       // (128, 4)
    const int  e4 = (int)(MD / 4 / 256);      // float4 elementwise blocks

    // one-time weight convert+transpose to bf16 [N][K]
    convT_bf16<<<(int)((size_t)D * Din / 256), blk, 0, stream>>>(W_embed, wtE, Din, D, (size_t)D * Din);
    for (int i = 0; i < S; ++i) {
        convT_bf16<<<(int)(DD / 256), blk, 0, stream>>>(Wq + i * DD, wtQ + i * DD, D, D, DD);
        convT_bf16<<<(int)(DD / 256), blk, 0, stream>>>(Wk + i * DD, wtK + i * DD, D, D, DD);
        convT_bf16<<<(int)(DD / 256), blk, 0, stream>>>(Wv + i * DD, wtV + i * DD, D, D, DD);
        convT_bf16<<<(int)(DD / 256), blk, 0, stream>>>(Wc1 + i * DD, wtC + i * DD, D, D, DD);
    }

    // embedding matmul + BN1
    gemm_wt<<<ggrid, blk, 0, stream>>>(x, wtE, nullptr, t0, M, D, Din, 0);
    bn_stats<<<D / 256, 256, 0, stream>>>(t0, nullptr, stats, M, D);
    bn_apply<<<e4, 256, 0, stream>>>(t0, nullptr, stats, bn1_g, bn1_b, enc, MD / 4);

    for (int i = 0; i < S; ++i) {
        // Q/K/V (Dense + relu)
        gemm_wt<<<ggrid, blk, 0, stream>>>(enc, wtQ + i * DD, bq + i * D, qb, M, D, D, 1);
        gemm_wt<<<ggrid, blk, 0, stream>>>(enc, wtK + i * DD, bk + i * D, kb, M, D, D, 1);
        gemm_wt<<<ggrid, blk, 0, stream>>>(enc, wtV + i * DD, bv + i * D, vb, M, D, D, 1);
        // attention
        flash_attn<<<agrid, blk, 0, stream>>>(qb, kb, vb, t0);
        // BN(attn_out + enc) -> enc   (residual fused into BN)
        bn_stats<<<D / 256, 256, 0, stream>>>(t0, enc, stats, M, D);
        bn_apply<<<e4, 256, 0, stream>>>(t0, enc, stats, bng + i * D, bnb + i * D, enc, MD / 4);
        // feedforward: conv1 applied twice (faithful to reference)
        gemm_wt<<<ggrid, blk, 0, stream>>>(enc, wtC + i * DD, bc1 + i * D, t0, M, D, D, 1);
        gemm_wt<<<ggrid, blk, 0, stream>>>(t0,  wtC + i * DD, bc1 + i * D, qb, M, D, D, 1);
        // BN(f2 + enc) -> enc (or d_out on the last layer)
        bn_stats<<<D / 256, 256, 0, stream>>>(qb, enc, stats, M, D);
        float* dst = (i == S - 1) ? out : enc;
        bn_apply<<<e4, 256, 0, stream>>>(qb, enc, stats, bn2g + i * D, bn2b + i * D, dst, MD / 4);
    }
}
